// TextAttention_3977139716516
// MI455X (gfx1250) — matrix operand order repaired and checked by reading, not compiled, not run on hardware
//
#include <hip/hip_runtime.h>

// ---------------------------------------------------------------------------
// Problem constants (fixed by the reference)
// ---------------------------------------------------------------------------
#define B_  32
#define S_  4096
#define D_  1024   // K dimension of big GEMM
#define H_  512    // N dimension of big GEMM
#define ROWS 64    // s-rows per workgroup in scores kernel (4 WMMA M-tiles)

typedef __attribute__((ext_vector_type(16))) __bf16 v16bf;
typedef __attribute__((ext_vector_type(8)))  float  v8f;

// round-to-nearest-even f32 -> bf16, packed pair into one dword (lo | hi<<16)
__device__ __forceinline__ unsigned int f2bf_pack(float lo, float hi) {
    unsigned int ul = __builtin_bit_cast(unsigned int, lo);
    unsigned int uh = __builtin_bit_cast(unsigned int, hi);
    unsigned int rl = (ul + 0x7FFFu + ((ul >> 16) & 1u)) >> 16;
    unsigned int rh = (uh + 0x7FFFu + ((uh >> 16) & 1u)) >> 16;
    return (rl & 0xFFFFu) | (rh << 16);
}

// ---------------------------------------------------------------------------
// Kernel 0: convert W1 (f32 [D,H] row-major) into bf16 packed in WMMA
// B-fragment order:  dst[(((kt*2 + g)*H_ + col)*8 + j]  (dwords)
//   kt  = K-tile (32 K values per WMMA)      0..31
//   g   = lane group (lanes 0-15 hold K 0..15, lanes 16-31 hold K 16..31)
//   col = N column                           0..511
//   j   = VGPR index; dword j = {K=k0+2j+1 , K=k0+2j},  k0 = kt*32 + g*16
// A lane later reads its full 16-half fragment as 32 contiguous bytes.
// ---------------------------------------------------------------------------
__global__ __launch_bounds__(256) void pack_w1_kernel(
    const float* __restrict__ W1, unsigned int* __restrict__ packB)
{
    unsigned int id = blockIdx.x * 256u + threadIdx.x;   // 0 .. 262143 dwords
    int j   = id & 7;
    int col = (id >> 3) & (H_ - 1);
    int g   = (id >> 12) & 1;
    int kt  = id >> 13;
    int k0  = kt * 32 + g * 16 + 2 * j;
    float lo = W1[(size_t)k0 * H_ + col];
    float hi = W1[(size_t)(k0 + 1) * H_ + col];
    packB[id] = f2bf_pack(lo, hi);
}

// ---------------------------------------------------------------------------
// Kernel 1: fused  scores[b,s] = w2 . relu( (text[b,s,:]*prev[b,:]) @ W1 + b1 ) + b2
// One workgroup = (batch b, 64 consecutive s rows). 8 waves, each owns 64 of
// the 512 H columns; 4 M-tiles x 4 N-tiles of f32 accumulators per wave, so
// every B fragment fetched from L2 feeds 4 WMMAs.
// A staged in LDS in contiguous-K fragment order, matching packed B:
//   lanes 0-15 : a[0..15]=K 0..15  (of the 32-K tile)
//   lanes 16-31: a[0..15]=K 16..31
// LDS layout: Apack[(kt*4 + rowGroup)*32 + lane][8 dwords]  == 128 KB.
// ---------------------------------------------------------------------------
__global__ __launch_bounds__(256) void scores_kernel(
    const float* __restrict__ text, const float* __restrict__ prev,
    const unsigned int* __restrict__ packB,
    const float* __restrict__ b1, const float* __restrict__ w2,
    const float* __restrict__ b2p, float* __restrict__ scores)
{
    __shared__ unsigned int Apack[32 * 4 * 32 * 8];   // 131072 bytes

    const int t  = threadIdx.x;
    const int b  = blockIdx.x >> 6;                   // S_/ROWS = 64 tiles
    const int s0 = (blockIdx.x & 63) * ROWS;

    const float* trow = text + ((size_t)b * S_ + s0) * D_;
    const float* pv   = prev + (size_t)b * D_;

    // ---- stage A: bf16(text * prev) into LDS in fragment order -------------
    #pragma unroll 4
    for (int it = 0; it < (ROWS * D_ / 2) / 256; ++it) {   // 128 iters
        int idx = t + 256 * it;            // pair index, 0..32767
        int r   = idx >> 9;                // s-row in tile, 0..63
        int kp  = idx & 511;
        int k   = kp * 2;                  // even K
        float2 tv = *(const float2*)(trow + (size_t)r * D_ + k);
        float  p0 = pv[k], p1 = pv[k + 1];
        unsigned int d = f2bf_pack(tv.x * p0, tv.y * p1);
        int kk = k & 31, kt = k >> 5;
        int g  = kk >> 4;                             // lane group
        int i  = kk & 15;                             // half index in frag
        int j  = i >> 1;                              // dword in frag
        int rg = r >> 4, rr = r & 15;
        int lane = g * 16 + rr;
        Apack[(((kt * 4 + rg) * 32) + lane) * 8 + j] = d;
    }
    __syncthreads();

    const int wave = t >> 5;
    const int lane = t & 31;
    const int g    = lane >> 4;     // lane group (selects K 0..15 vs 16..31)
    const int ln   = lane & 15;     // N within tile / M-row selector
    const int c0   = wave * 64;     // this wave's column block

    v8f acc[4][4];
    #pragma unroll
    for (int m = 0; m < 4; ++m)
        #pragma unroll
        for (int n = 0; n < 4; ++n)
            acc[m][n] = (v8f){0.f,0.f,0.f,0.f,0.f,0.f,0.f,0.f};

    // ---- main WMMA loop: K = 1024 in 32-wide steps -------------------------
    for (int kt = 0; kt < 32; ++kt) {
        v16bf a0 = *(const v16bf*)&Apack[((kt * 4 + 0) * 32 + lane) * 8];
        v16bf a1 = *(const v16bf*)&Apack[((kt * 4 + 1) * 32 + lane) * 8];
        v16bf a2 = *(const v16bf*)&Apack[((kt * 4 + 2) * 32 + lane) * 8];
        v16bf a3 = *(const v16bf*)&Apack[((kt * 4 + 3) * 32 + lane) * 8];
        const unsigned int* bb = packB + ((size_t)(kt * 2 + g) * H_) * 8;
        #pragma unroll
        for (int nt = 0; nt < 4; ++nt) {
            int col = c0 + nt * 16 + ln;
            v16bf bf = *(const v16bf*)(bb + col * 8);
            acc[0][nt] = __builtin_amdgcn_wmma_f32_16x16x32_bf16(
                false, a0, false, bf, (short)0, acc[0][nt], false, false);
            acc[1][nt] = __builtin_amdgcn_wmma_f32_16x16x32_bf16(
                false, a1, false, bf, (short)0, acc[1][nt], false, false);
            acc[2][nt] = __builtin_amdgcn_wmma_f32_16x16x32_bf16(
                false, a2, false, bf, (short)0, acc[2][nt], false, false);
            acc[3][nt] = __builtin_amdgcn_wmma_f32_16x16x32_bf16(
                false, a3, false, bf, (short)0, acc[3][nt], false, false);
        }
    }

    // ---- fused epilogue: +b1, ReLU, dot with w2 ----------------------------
    // C layout: element (M,N): lane = N + 16*(M/8), vgpr = M%8.
    float ps[4][8];
    #pragma unroll
    for (int m = 0; m < 4; ++m)
        #pragma unroll
        for (int v = 0; v < 8; ++v)
            ps[m][v] = 0.f;

    #pragma unroll
    for (int nt = 0; nt < 4; ++nt) {
        int col = c0 + nt * 16 + ln;
        float bb1 = b1[col];
        float ww  = w2[col];
        #pragma unroll
        for (int m = 0; m < 4; ++m)
            #pragma unroll
            for (int v = 0; v < 8; ++v) {
                float h = fmaxf(acc[m][nt][v] + bb1, 0.f);
                ps[m][v] += h * ww;
            }
    }
    // reduce over the 16 lanes of each half-wave (columns of this wave)
    #pragma unroll
    for (int off = 1; off <= 8; off <<= 1) {
        #pragma unroll
        for (int m = 0; m < 4; ++m)
            #pragma unroll
            for (int v = 0; v < 8; ++v)
                ps[m][v] += __shfl_xor(ps[m][v], off, 32);
    }

    __syncthreads();                    // all LDS A reads complete
    float* scoreAcc = (float*)Apack;    // reuse first 256B of LDS
    if (t < ROWS) scoreAcc[t] = 0.f;
    __syncthreads();

    if (ln == 0) {                      // lanes 0 (rows m%16<8) and 16 (>=8)
        int rbase = g * 8;
        #pragma unroll
        for (int m = 0; m < 4; ++m)
            #pragma unroll
            for (int v = 0; v < 8; ++v)
                atomicAdd(&scoreAcc[m * 16 + rbase + v], ps[m][v]);
    }
    __syncthreads();

    if (t < ROWS)
        scores[(size_t)b * S_ + s0 + t] = scoreAcc[t] + b2p[0];
}

// ---------------------------------------------------------------------------
// Kernel 2: softmax over S per batch -> weights
// ---------------------------------------------------------------------------
__global__ __launch_bounds__(256) void softmax_kernel(
    const float* __restrict__ scores, float* __restrict__ weights)
{
    __shared__ float red[256];
    const int b = blockIdx.x, t = threadIdx.x;
    const float* sc = scores + (size_t)b * S_;

    float loc[S_ / 256];
    float mx = -__builtin_inff();
    #pragma unroll
    for (int i = 0; i < S_ / 256; ++i) {
        loc[i] = sc[t + 256 * i];
        mx = fmaxf(mx, loc[i]);
    }
    red[t] = mx; __syncthreads();
    for (int o = 128; o > 0; o >>= 1) {
        if (t < o) red[t] = fmaxf(red[t], red[t + o]);
        __syncthreads();
    }
    mx = red[0]; __syncthreads();

    float sum = 0.f;
    #pragma unroll
    for (int i = 0; i < S_ / 256; ++i) {
        loc[i] = __expf(loc[i] - mx);
        sum += loc[i];
    }
    red[t] = sum; __syncthreads();
    for (int o = 128; o > 0; o >>= 1) {
        if (t < o) red[t] += red[t + o];
        __syncthreads();
    }
    float inv = 1.f / red[0];
    #pragma unroll
    for (int i = 0; i < S_ / 256; ++i)
        weights[(size_t)b * S_ + t + 256 * i] = loc[i] * inv;
}

// ---------------------------------------------------------------------------
// Kernel 3: out[b,d] = sum_s text[b,s,d] * weights[b,s]
// grid = B * (D/256); weights staged through LDS.
// ---------------------------------------------------------------------------
__global__ __launch_bounds__(256) void outsum_kernel(
    const float* __restrict__ text, const float* __restrict__ weights,
    float* __restrict__ out)
{
    __shared__ float wsh[256];
    const int b  = blockIdx.x >> 2;
    const int c0 = (blockIdx.x & 3) * 256;
    const int t  = threadIdx.x;

    const float* tb = text + (size_t)b * S_ * D_ + c0 + t;
    const float* wb = weights + (size_t)b * S_;

    float acc = 0.f;
    for (int s0 = 0; s0 < S_; s0 += 256) {
        __syncthreads();
        wsh[t] = wb[s0 + t];
        __syncthreads();
        #pragma unroll 8
        for (int ss = 0; ss < 256; ++ss)
            acc += tb[(size_t)(s0 + ss) * D_] * wsh[ss];
    }
    out[(size_t)b * D_ + c0 + t] = acc;
}

// ---------------------------------------------------------------------------
// Host launcher. ws layout: [0,1MB) packed bf16 W1 ; [1MB,1.5MB) scores ;
// [1.5MB,2MB) softmax weights.
// ---------------------------------------------------------------------------
extern "C" void kernel_launch(void* const* d_in, const int* in_sizes, int n_in,
                              void* d_out, int out_size, void* d_ws, size_t ws_size,
                              hipStream_t stream)
{
    const float* text = (const float*)d_in[0];
    const float* prev = (const float*)d_in[1];
    const float* W1   = (const float*)d_in[2];
    const float* b1   = (const float*)d_in[3];
    const float* w2   = (const float*)d_in[4];
    const float* b2   = (const float*)d_in[5];
    float* out = (float*)d_out;

    unsigned int* packB  = (unsigned int*)d_ws;                         // 1 MB
    float* scores  = (float*)((char*)d_ws + (1u << 20));                // 512 KB
    float* weights = (float*)((char*)d_ws + (1u << 20) + (512u << 10)); // 512 KB

    pack_w1_kernel<<<(D_ * H_ / 2) / 256, 256, 0, stream>>>(W1, packB);
    scores_kernel<<<B_ * (S_ / ROWS), 256, 0, stream>>>(
        text, prev, packB, b1, w2, b2, scores);
    softmax_kernel<<<B_, 256, 0, stream>>>(scores, weights);
    outsum_kernel<<<B_ * (D_ / 256), 256, 0, stream>>>(text, weights, out);
}
